// GPT_32530082300550
// MI455X (gfx1250) — matrix operand order repaired and checked by reading, not compiled, not run on hardware
//
#include <hip/hip_runtime.h>
#include <hip/hip_bf16.h>
#include <math.h>

// GPT-2-ish forward on gfx1250 (MI455X), all matmuls via v_wmma_f32_16x16x32_f16.
// Sizes (compile-time): L=12, D=1024, H=16, dh=64, V=32000, B=2, T=1024.

#define Dm 1024
#define Tm 1024
#define Bbatch 2
#define Hm 16
#define Lm 12
#define Vm 32000

typedef __attribute__((ext_vector_type(16))) _Float16 v16h;
typedef __attribute__((ext_vector_type(8)))  float    v8f;

union Frag32 {            // one 16x32 (A) or 32x16 (B) f16 fragment = 32 bytes/lane
  uint4     q[2];
  v16h      h;
  _Float16  e[16];
};

__device__ __forceinline__ v8f wmma_f32_f16(const Frag32& a, const Frag32& b, v8f c) {
  return __builtin_amdgcn_wmma_f32_16x16x32_f16(false, a.h, false, b.h, (short)0, c,
                                                false, false);
}

__device__ __forceinline__ float gelu_exact(float v) {
  return 0.5f * v * (1.0f + erff(v * 0.70710678118654752f));
}

// ---------------------------------------------------------------------------
// Embedding: x[row,:] = wte[tok[row],:] + wpe[row%T,:]
// ---------------------------------------------------------------------------
__global__ __launch_bounds__(256) void embed_kernel(const int* __restrict__ tokens,
                                                    const float* __restrict__ wte,
                                                    const float* __restrict__ wpe,
                                                    float* __restrict__ x) {
  const int row = blockIdx.x;
  const int t   = row & (Tm - 1);
  const int tok = tokens[row];
#pragma unroll
  for (int i = 0; i < 4; ++i) {
    const int c = threadIdx.x + i * 256;
    x[(size_t)row * Dm + c] = wte[(size_t)tok * Dm + c] + wpe[(size_t)t * Dm + c];
  }
}

// ---------------------------------------------------------------------------
// LayerNorm, one 256-thread block per row of D=1024.
// ---------------------------------------------------------------------------
__global__ __launch_bounds__(256) void ln_kernel(const float* __restrict__ x,
                                                 const float* __restrict__ w,
                                                 const float* __restrict__ b,
                                                 float* __restrict__ out) {
  __shared__ float psum[8], psq[8], stats[2];
  const int tid  = threadIdx.x;
  const int lane = tid & 31;
  const int wid  = tid >> 5;
  const float* xr = x + (size_t)blockIdx.x * Dm;

  float v[4], s = 0.f, ss = 0.f;
#pragma unroll
  for (int i = 0; i < 4; ++i) {
    v[i] = xr[tid + i * 256];
    s  += v[i];
    ss += v[i] * v[i];
  }
#pragma unroll
  for (int off = 1; off < 32; off <<= 1) {
    s  += __shfl_xor(s,  off, 32);
    ss += __shfl_xor(ss, off, 32);
  }
  if (lane == 0) { psum[wid] = s; psq[wid] = ss; }
  __syncthreads();
  if (tid == 0) {
    float S = 0.f, SS = 0.f;
    for (int i = 0; i < 8; ++i) { S += psum[i]; SS += psq[i]; }
    const float mu  = S / (float)Dm;
    const float var = SS / (float)Dm - mu * mu;
    stats[0] = mu;
    stats[1] = rsqrtf(var + 1e-5f);
  }
  __syncthreads();
  const float mu = stats[0], rstd = stats[1];
#pragma unroll
  for (int i = 0; i < 4; ++i) {
    const int c = tid + i * 256;
    out[(size_t)blockIdx.x * Dm + c] = (v[i] - mu) * rstd * w[c] + b[c];
  }
}

// ---------------------------------------------------------------------------
// WMMA GEMM: C[M,N] = act(A[M,K] @ W[K,N] + bias) + res
//   Block tile 128x256, 8 waves (2x4), wave tile 64x64 (4x4 WMMA tiles), BK=32.
//   Double-buffered LDS: next tile's global loads issue BEFORE this tile's
//   WMMAs; f32->f16 conversion + ds_store happen AFTER, so HBM/L2 latency is
//   hidden behind 16 WMMAs. One barrier per k-step.
//   act: 0 = none, 1 = exact GELU.  bias/res may be nullptr.
// ---------------------------------------------------------------------------
#define BM 128
#define BN 256
#define BK 32
#define LDSPITCH 40  // 32 + 8 halfs padding; 80B keeps 16B alignment for b128 reads

__global__ __launch_bounds__(256) void wmma_gemm_kernel(
    const float* __restrict__ A, const float* __restrict__ Bw,
    const float* __restrict__ bias, const float* __restrict__ res,
    float* __restrict__ C, int M, int N, int K, int act) {
  __shared__ _Float16 As[2][BM][LDSPITCH];   // row-major [m][k]
  __shared__ _Float16 Bs[2][BN][LDSPITCH];   // K-transposed [n][k]

  const int tid  = threadIdx.x;
  const int lane = tid & 31;
  const int wid  = tid >> 5;
  const int wm   = wid & 1;       // 2 waves along M
  const int wn   = wid >> 1;      // 4 waves along N
  const int half = lane >> 4;
  const int l16  = lane & 15;
  const int bm   = blockIdx.y * BM;
  const int bn   = blockIdx.x * BN;

  // per-thread staging coordinates (computed once)
  int rA[4], kqA[4], kkB[8], nB[8];
#pragma unroll
  for (int i = 0; i < 4; ++i) {
    const int idx = tid + i * 256;          // 0..1023 over 128x8 float4s
    rA[i]  = idx >> 3;
    kqA[i] = (idx & 7) * 4;
  }
#pragma unroll
  for (int i = 0; i < 8; ++i) {
    const int idx = tid + i * 256;          // 0..2047 over 32x64 float4s
    kkB[i] = idx >> 6;
    nB[i]  = (idx & 63) * 4;
  }

  float4 a4[4], b4[8];
  // ---- load tile kt into registers (12 independent global_load_b128) ----
  auto load_tile = [&](int kt) {
#pragma unroll
    for (int i = 0; i < 4; ++i)
      a4[i] = *reinterpret_cast<const float4*>(A + (size_t)(bm + rA[i]) * K + kt + kqA[i]);
#pragma unroll
    for (int i = 0; i < 8; ++i)
      b4[i] = *reinterpret_cast<const float4*>(Bw + (size_t)(kt + kkB[i]) * N + bn + nB[i]);
  };
  // ---- convert f32->f16 and store registers into LDS buffer ----
  auto store_tile = [&](int buf) {
#pragma unroll
    for (int i = 0; i < 4; ++i) {
      _Float16* dst = &As[buf][rA[i]][kqA[i]];
      dst[0] = (_Float16)a4[i].x; dst[1] = (_Float16)a4[i].y;
      dst[2] = (_Float16)a4[i].z; dst[3] = (_Float16)a4[i].w;
    }
#pragma unroll
    for (int i = 0; i < 8; ++i) {
      Bs[buf][nB[i] + 0][kkB[i]] = (_Float16)b4[i].x;
      Bs[buf][nB[i] + 1][kkB[i]] = (_Float16)b4[i].y;
      Bs[buf][nB[i] + 2][kkB[i]] = (_Float16)b4[i].z;
      Bs[buf][nB[i] + 3][kkB[i]] = (_Float16)b4[i].w;
    }
  };

  v8f zero = {};
  v8f acc[4][4];
#pragma unroll
  for (int mi = 0; mi < 4; ++mi)
#pragma unroll
    for (int ni = 0; ni < 4; ++ni) acc[mi][ni] = zero;

  load_tile(0);
  store_tile(0);
  __syncthreads();

  int buf = 0;
  for (int kt = 0; kt < K; kt += BK) {
    const bool has_next = (kt + BK) < K;
    if (has_next) load_tile(kt + BK);          // issue early, consume after WMMAs
    if (kt + 2 * BK < K) {                     // L2 prefetch two tiles ahead
      __builtin_prefetch(A + (size_t)(bm + (tid >> 1)) * K + kt + 2 * BK, 0, 1);
      __builtin_prefetch(Bw + (size_t)(kt + 2 * BK + (tid >> 3)) * N + bn, 0, 1);
    }

    // ---- compute from LDS buffer `buf` ----
    Frag32 bf[4];
#pragma unroll
    for (int ni = 0; ni < 4; ++ni) {
      const _Float16* p = &Bs[buf][wn * 64 + ni * 16 + l16][half * 16];
      bf[ni].q[0] = *reinterpret_cast<const uint4*>(p);
      bf[ni].q[1] = *reinterpret_cast<const uint4*>(p + 8);
    }
#pragma unroll
    for (int mi = 0; mi < 4; ++mi) {
      Frag32 af;
      const _Float16* p = &As[buf][wm * 64 + mi * 16 + l16][half * 16];
      af.q[0] = *reinterpret_cast<const uint4*>(p);
      af.q[1] = *reinterpret_cast<const uint4*>(p + 8);
#pragma unroll
      for (int ni = 0; ni < 4; ++ni)
        acc[mi][ni] = wmma_f32_f16(af, bf[ni], acc[mi][ni]);
    }

    if (has_next) store_tile(buf ^ 1);         // cvt+ds_store after compute
    __syncthreads();
    buf ^= 1;
  }

  // ---- epilogue: bias -> act -> residual -> store ----
#pragma unroll
  for (int ni = 0; ni < 4; ++ni) {
    const int col = bn + wn * 64 + ni * 16 + l16;
    const float bv = bias ? bias[col] : 0.0f;
#pragma unroll
    for (int mi = 0; mi < 4; ++mi) {
      const int row0 = bm + wm * 64 + mi * 16 + half * 8;
#pragma unroll
      for (int j = 0; j < 8; ++j) {
        float v = acc[mi][ni][j] + bv;
        if (act == 1) v = gelu_exact(v);
        const size_t off = (size_t)(row0 + j) * N + col;
        if (res) v += res[off];
        C[off] = v;
      }
    }
  }
}

// ---------------------------------------------------------------------------
// Flash-style causal attention with WMMA.
// One wave per (batch b, head h, 16-query tile qt). qkv: [B*T, 3D] fp32.
//   S(16q x 16k) = (Q/8) K^T   via 2 WMMAs over dh=64
//   online softmax (16-lane shuffle row-reductions in C layout)
//   P bounced through LDS (C layout -> A fragment layout), O += P V (4 WMMAs)
// ---------------------------------------------------------------------------
__global__ __launch_bounds__(32) void attn_kernel(const float* __restrict__ qkv,
                                                  float* __restrict__ y) {
  __shared__ _Float16 Ps[16][LDSPITCH];
  const int lane = threadIdx.x & 31;
  const int half = lane >> 4;
  const int l16  = lane & 15;
  const int bid  = blockIdx.x;           // B*H*(T/16) blocks
  const int qt   = bid & 63;             // T/16 = 64
  const int h    = (bid >> 6) & 15;
  const int b    = bid >> 10;
  const int ldq  = 3 * Dm;
  const float* base = qkv + (size_t)b * Tm * ldq;

  // Q fragments (A layout: M = l16, K pieces by half), prescaled by 1/sqrt(64)
  Frag32 qa[2];
  {
    const float* qr = base + (size_t)(qt * 16 + l16) * ldq + h * 64;
#pragma unroll
    for (int c = 0; c < 2; ++c) {
#pragma unroll
      for (int j = 0; j < 8; ++j) {
        const int kk = half * 16 + 2 * j;
        const float2 f = *reinterpret_cast<const float2*>(qr + c * 32 + kk);
        qa[c].e[2 * j]     = (_Float16)(f.x * 0.125f);
        qa[c].e[2 * j + 1] = (_Float16)(f.y * 0.125f);
      }
    }
  }

  v8f zero = {};
  v8f o[4];
#pragma unroll
  for (int n = 0; n < 4; ++n) o[n] = zero;
  float m[8], lsum[8];
#pragma unroll
  for (int j = 0; j < 8; ++j) { m[j] = -1e30f; lsum[j] = 0.f; }

  const int kmax = qt * 16 + 15;
  for (int kc = 0; kc * 32 <= kmax; ++kc) {
    const int kbase = kc * 32;

    // ---- scores: two 16-key tiles, reduction over dh=64 (2 WMMAs each) ----
    v8f s[2];
#pragma unroll
    for (int t = 0; t < 2; ++t) {
      v8f ssum = zero;
#pragma unroll
      for (int c = 0; c < 2; ++c) {
        Frag32 kb;  // B layout: n = l16 (key), VGPR j holds d-pair half*16+2j
        const float* kr =
            base + (size_t)(kbase + t * 16 + l16) * ldq + Dm + h * 64 + c * 32;
#pragma unroll
        for (int j = 0; j < 8; ++j) {
          const float2 f = *reinterpret_cast<const float2*>(kr + half * 16 + 2 * j);
          kb.e[2 * j]     = (_Float16)f.x;
          kb.e[2 * j + 1] = (_Float16)f.y;
        }
        ssum = wmma_f32_f16(qa[c], kb, ssum);
      }
      s[t] = ssum;
    }

    // ---- causal mask + online softmax (row M = j + 8*half, col = l16) ----
    float p0[8], p1[8], alpha[8];
#pragma unroll
    for (int j = 0; j < 8; ++j) {
      const int qi = qt * 16 + j + 8 * half;
      float v0 = s[0][j]; if (kbase + l16 > qi)       v0 = -1e30f;
      float v1 = s[1][j]; if (kbase + 16 + l16 > qi)  v1 = -1e30f;
      float mx = fmaxf(v0, v1);
#pragma unroll
      for (int off = 1; off < 16; off <<= 1) mx = fmaxf(mx, __shfl_xor(mx, off, 16));
      const float mnew = fmaxf(m[j], mx);
      v0 = __expf(v0 - mnew);
      v1 = __expf(v1 - mnew);
      float rs = v0 + v1;
#pragma unroll
      for (int off = 1; off < 16; off <<= 1) rs += __shfl_xor(rs, off, 16);
      alpha[j] = __expf(m[j] - mnew);
      lsum[j]  = lsum[j] * alpha[j] + rs;
      m[j]     = mnew;
      p0[j] = v0; p1[j] = v1;
    }
#pragma unroll
    for (int n = 0; n < 4; ++n)
#pragma unroll
      for (int j = 0; j < 8; ++j) o[n][j] *= alpha[j];

    // ---- P: C layout -> LDS -> A fragment ----
    __syncthreads();
#pragma unroll
    for (int j = 0; j < 8; ++j) {
      Ps[j + 8 * half][l16]      = (_Float16)p0[j];
      Ps[j + 8 * half][16 + l16] = (_Float16)p1[j];
    }
    __syncthreads();
    Frag32 pa;
    {
      const _Float16* pp = &Ps[l16][half * 16];
      pa.q[0] = *reinterpret_cast<const uint4*>(pp);
      pa.q[1] = *reinterpret_cast<const uint4*>(pp + 8);
    }

    // ---- O += P @ V : 4 output column tiles of 16 ----
#pragma unroll
    for (int n = 0; n < 4; ++n) {
      Frag32 vb;  // B layout: n = l16 (dim), VGPR j holds key-pair half*16+2j
      const float* vr =
          base + (size_t)kbase * ldq + 2 * Dm + h * 64 + n * 16 + l16;
#pragma unroll
      for (int j = 0; j < 8; ++j) {
        const int key = half * 16 + 2 * j;
        vb.e[2 * j]     = (_Float16)vr[(size_t)key * ldq];
        vb.e[2 * j + 1] = (_Float16)vr[(size_t)(key + 1) * ldq];
      }
      o[n] = wmma_f32_f16(pa, vb, o[n]);
    }
  }

  // ---- normalize and write y[b*T + q, h*64 + n16*16 + l16] ----
#pragma unroll
  for (int n = 0; n < 4; ++n)
#pragma unroll
    for (int j = 0; j < 8; ++j) {
      const int row = b * Tm + qt * 16 + j + 8 * half;
      y[(size_t)row * Dm + h * 64 + n * 16 + l16] = o[n][j] / lsum[j];
    }
}

// ---------------------------------------------------------------------------
// Driver
// ---------------------------------------------------------------------------
extern "C" void kernel_launch(void* const* d_in, const int* in_sizes, int n_in,
                              void* d_out, int out_size, void* d_ws, size_t ws_size,
                              hipStream_t stream) {
  (void)in_sizes; (void)n_in; (void)out_size; (void)ws_size;
  const int*   tokens = (const int*)  d_in[0];
  const float* wte    = (const float*)d_in[1];
  const float* wpe    = (const float*)d_in[2];
  const float* ln1_w  = (const float*)d_in[3];
  const float* ln1_b  = (const float*)d_in[4];
  const float* ln2_w  = (const float*)d_in[5];
  const float* ln2_b  = (const float*)d_in[6];
  const float* w_qkv  = (const float*)d_in[7];
  const float* b_qkv  = (const float*)d_in[8];
  const float* w_proj = (const float*)d_in[9];
  const float* b_proj = (const float*)d_in[10];
  const float* w_fc1  = (const float*)d_in[11];
  const float* b_fc1  = (const float*)d_in[12];
  const float* w_fc2  = (const float*)d_in[13];
  const float* b_fc2  = (const float*)d_in[14];
  const float* lnf_w  = (const float*)d_in[15];
  const float* lnf_b  = (const float*)d_in[16];
  const float* w_out  = (const float*)d_in[17];

  const size_t NT = (size_t)Bbatch * Tm;       // 2048 token rows
  char* ws = (char*)d_ws;
  float* x   = (float*)(ws);                                      //  8 MB
  float* hln = (float*)(ws + NT * Dm * 4);                        //  8 MB
  float* qkv = (float*)(ws + 2 * NT * Dm * 4);                    // 24 MB
  float* yb  = (float*)(ws + 2 * NT * Dm * 4 + NT * 3 * Dm * 4);  //  8 MB
  float* h3  = (float*)(ws + 3 * NT * Dm * 4 + NT * 3 * Dm * 4);  //  8 MB

  const dim3 blk(256);
  embed_kernel<<<(int)NT, blk, 0, stream>>>(tokens, wte, wpe, x);

  for (int l = 0; l < Lm; ++l) {
    ln_kernel<<<(int)NT, blk, 0, stream>>>(x, ln1_w + l * Dm, ln1_b + l * Dm, hln);
    wmma_gemm_kernel<<<dim3(3 * Dm / BN, NT / BM), blk, 0, stream>>>(
        hln, w_qkv + (size_t)l * Dm * 3 * Dm, b_qkv + (size_t)l * 3 * Dm,
        nullptr, qkv, (int)NT, 3 * Dm, Dm, 0);
    attn_kernel<<<Bbatch * Hm * (Tm / 16), dim3(32), 0, stream>>>(qkv, yb);
    wmma_gemm_kernel<<<dim3(Dm / BN, NT / BM), blk, 0, stream>>>(
        yb, w_proj + (size_t)l * Dm * Dm, b_proj + (size_t)l * Dm,
        x, x, (int)NT, Dm, Dm, 0);
    ln_kernel<<<(int)NT, blk, 0, stream>>>(x, ln2_w + l * Dm, ln2_b + l * Dm, hln);
    wmma_gemm_kernel<<<dim3(Dm / BN, NT / BM), blk, 0, stream>>>(
        hln, w_fc1 + (size_t)l * Dm * Dm, b_fc1 + (size_t)l * Dm,
        nullptr, h3, (int)NT, Dm, Dm, 1 /*GELU*/);
    wmma_gemm_kernel<<<dim3(Dm / BN, NT / BM), blk, 0, stream>>>(
        h3, w_fc2 + (size_t)l * Dm * Dm, b_fc2 + (size_t)l * Dm,
        x, x, (int)NT, Dm, Dm, 0);
  }

  ln_kernel<<<(int)NT, blk, 0, stream>>>(x, lnf_w, lnf_b, hln);
  wmma_gemm_kernel<<<dim3(Vm / BN, NT / BM), blk, 0, stream>>>(
      hln, w_out, nullptr, nullptr, (float*)d_out, (int)NT, Vm, Dm, 0);
}
